// MMTs_43104291783485
// MI455X (gfx1250) — matrix operand order repaired and checked by reading, not compiled, not run on hardware
//
#include <hip/hip_runtime.h>

// ---------------------------------------------------------------------------
// Problem constants (from reference): B=2, M=4, S=1024, D=128, H=8, E=H*D=1024
// ---------------------------------------------------------------------------
#define S_LEN   1024
#define D_DIM   128
#define H_NUM   8
#define E_DIM   1024
#define BM_CNT  8                     // B*M
#define SCALE_F 0.08838834764831845f  // 1/sqrt(128)

typedef __bf16 bf16_t;
typedef __attribute__((ext_vector_type(16))) __bf16 v16bf;
typedef __attribute__((ext_vector_type(8)))  float  v8f;

// ---------------------------------------------------------------------------
// WMMA fragment loaders (CDNA5 ISA 7.12.2 layouts, wave32)
//  A (16x32 bf16): lane l(0..15) holds row M=l; half hi = lane>>4 selects
//    K octets: elems 0..7 -> K = hi*8+0..7, elems 8..15 -> K = 16+hi*8+0..7
//  B (32x16 bf16): lane l = column N; same K octets as A:
//    elems 0..7 -> K = hi*8+0..7, elems 8..15 -> K = 16+hi*8+0..7
//    from an N-major ("transposed") buffer: baseT[l*ld + K]
//  C/D (16x16 f32): VGPR r holds (M = r + hi*8, N = l)
// ---------------------------------------------------------------------------
__device__ __forceinline__ v16bf frag_a(const bf16_t* base, int ld, int lane) {
  const int l = lane & 15, hi = lane >> 4;
  const bf16_t* row = base + (size_t)l * ld;
  v16bf a;
#pragma unroll
  for (int j = 0; j < 8; ++j) a[j] = row[hi * 8 + j];
#pragma unroll
  for (int j = 0; j < 8; ++j) a[8 + j] = row[16 + hi * 8 + j];
  return a;
}

__device__ __forceinline__ v16bf frag_bt(const bf16_t* baseT, int ld, int lane) {
  const int l = lane & 15, hi = lane >> 4;
  const bf16_t* col = baseT + (size_t)l * ld;
  v16bf b;
#pragma unroll
  for (int e = 0; e < 8; ++e) b[e] = col[hi * 8 + e];
#pragma unroll
  for (int e = 0; e < 8; ++e) b[8 + e] = col[16 + hi * 8 + e];
  return b;
}

__device__ __forceinline__ v8f wmma_bf16(v16bf a, v16bf b, v8f c) {
  return __builtin_amdgcn_wmma_f32_16x16x32_bf16(
      /*neg_a=*/false, a, /*neg_b=*/false, b,
      /*c_mod=*/(short)0, c, /*reuse_a=*/false, /*reuse_b=*/false);
}

// ---------------------------------------------------------------------------
// Kernel 1: projection GEMM  Y = X @ W  (per b,m; per proj q/k/v)
//   X: [S, D] f32, W: [D, E] f32, Y stored bf16 head-major: [BM, H, S, D]
// Block: 256 thr (8 waves). Tile: 64 rows x 128 cols (cols == one head).
// Each wave: 32x32 (2x2 WMMA frags). K-loop: D=128 in 4 steps of 32.
// grid = (E/128=8 (=head), S/64=16, 3*BM=24)
// ---------------------------------------------------------------------------
__global__ __launch_bounds__(256) void mhsa_proj_kernel(
    const float* __restrict__ xq, const float* __restrict__ xk,
    const float* __restrict__ xv, const float* __restrict__ Wq,
    const float* __restrict__ Wk, const float* __restrict__ Wv,
    bf16_t* __restrict__ Qh, bf16_t* __restrict__ Kh, bf16_t* __restrict__ Vh) {
  __shared__ bf16_t sA[64][32];    // X tile  (rows x K)            4 KB
  __shared__ bf16_t sBt[128][32];  // W tile, N-major (cols x K)    8 KB

  const int head = blockIdx.x;          // 0..7  -> e0 = head*128
  const int s0   = blockIdx.y * 64;
  const int bz   = blockIdx.z;
  const int proj = bz >> 3;             // 0=q,1=k,2=v
  const int bm   = bz & 7;              // b*M + m
  const int m    = bm & 3;

  const float* X = (proj == 0 ? xq : proj == 1 ? xk : xv) + (size_t)bm * S_LEN * D_DIM;
  const float* W = (proj == 0 ? Wq : proj == 1 ? Wk : Wv) + (size_t)m * D_DIM * E_DIM;
  bf16_t* dst = (proj == 0 ? Qh : proj == 1 ? Kh : Vh) +
                ((size_t)(bm * H_NUM + head)) * S_LEN * D_DIM;

  const int tid  = threadIdx.x;
  const int w    = tid >> 5, lane = tid & 31;
  const int l    = lane & 15, hi = lane >> 4;
  const int wr   = w >> 2;    // 0..1 : row 32-block
  const int wc   = w & 3;     // 0..3 : col 32-block

  v8f c[2][2];
#pragma unroll
  for (int ra = 0; ra < 2; ++ra)
#pragma unroll
    for (int cb = 0; cb < 2; ++cb)
#pragma unroll
      for (int r = 0; r < 8; ++r) c[ra][cb][r] = 0.0f;

  for (int ks = 0; ks < 4; ++ks) {
    const int k0 = ks * 32;
    __syncthreads();
    // stage X tile (64x32): 8 floats/thread via 2x b128 loads -> 1x b128 store
    {
      int r = tid >> 2, kq = (tid & 3) * 8;
      const float* src = &X[(size_t)(s0 + r) * D_DIM + k0 + kq];
      float4 f0 = *(const float4*)src;
      float4 f1 = *(const float4*)(src + 4);
      alignas(16) bf16_t tmp[8] = {
          (bf16_t)f0.x, (bf16_t)f0.y, (bf16_t)f0.z, (bf16_t)f0.w,
          (bf16_t)f1.x, (bf16_t)f1.y, (bf16_t)f1.z, (bf16_t)f1.w};
      *(uint4*)&sA[r][kq] = *(const uint4*)tmp;
    }
    // stage W tile (32x128) transposed: b128 global reads, 2B LDS scatter
#pragma unroll
    for (int p = 0; p < 4; ++p) {
      int qi = tid + p * 256;          // 0..1023 float-quads
      int kk = qi >> 5, nq = (qi & 31) * 4;
      float4 f = *(const float4*)&W[(size_t)(k0 + kk) * E_DIM + head * 128 + nq];
      sBt[nq + 0][kk] = (bf16_t)f.x;
      sBt[nq + 1][kk] = (bf16_t)f.y;
      sBt[nq + 2][kk] = (bf16_t)f.z;
      sBt[nq + 3][kk] = (bf16_t)f.w;
    }
    __syncthreads();

#pragma unroll
    for (int ra = 0; ra < 2; ++ra) {
      v16bf a = frag_a(&sA[wr * 32 + ra * 16][0], 32, lane);
#pragma unroll
      for (int cb = 0; cb < 2; ++cb) {
        v16bf b = frag_bt(&sBt[wc * 32 + cb * 16][0], 32, lane);
        c[ra][cb] = wmma_bf16(a, b, c[ra][cb]);
      }
    }
  }

  // store bf16, head-major [s, d] (cols of this block == dims of head `head`)
#pragma unroll
  for (int ra = 0; ra < 2; ++ra)
#pragma unroll
    for (int cb = 0; cb < 2; ++cb)
#pragma unroll
      for (int r = 0; r < 8; ++r) {
        int s = s0 + wr * 32 + ra * 16 + r + hi * 8;
        int d = wc * 32 + cb * 16 + l;
        dst[(size_t)s * D_DIM + d] = (bf16_t)c[ra][cb][r];
      }
}

// ---------------------------------------------------------------------------
// Kernel 2: fused flash attention + head-mean + residual.
// Block: 256 thr (8 waves), owns one (bm, 64-row q tile); loops h=0..7 and
// 16 key tiles of 64. K tile is staged with GLOBAL_LOAD_ASYNC_TO_LDS_B128
// (ASYNCcnt-tracked DMA copy, no VGPR round-trip); V tile is read b128 and
// transposed with 2B LDS stores; next tile is prefetched (global_prefetch_b8).
// grid = (S/64=16, BM=8)
// ---------------------------------------------------------------------------
__global__ __launch_bounds__(256) void mhsa_attn_kernel(
    const bf16_t* __restrict__ Qh, const bf16_t* __restrict__ Kh,
    const bf16_t* __restrict__ Vh, const float* __restrict__ qin,
    float* __restrict__ out) {
  __shared__ bf16_t sK[64][128];   // key tile, row-major       16 KB
  __shared__ bf16_t sVt[128][64];  // value tile, d-major       16 KB
  __shared__ float  sS[64][68];    // scores (padded)           17 KB
  __shared__ bf16_t sP[64][64];    // exp(scores) bf16           8 KB
  __shared__ float  sM[64], sL[64], sCorr[64];
  __shared__ float  sRed[64][4];

  const int tid  = threadIdx.x;
  const int w    = tid >> 5, lane = tid & 31;
  const int l    = lane & 15, hi = lane >> 4;
  const int qb   = w >> 1;  // 0..3
  const int kh   = w & 1;   // score-phase key half
  const int dh   = w & 1;   // PV-phase dim half
  const int s0   = blockIdx.x * 64;
  const int bm   = blockIdx.y;

  v8f macc[4];
#pragma unroll
  for (int cb = 0; cb < 4; ++cb)
#pragma unroll
    for (int r = 0; r < 8; ++r) macc[cb][r] = 0.0f;

  for (int h = 0; h < H_NUM; ++h) {
    const size_t hoff = ((size_t)(bm * H_NUM + h)) * S_LEN * D_DIM;
    const bf16_t* Qg = Qh + hoff;
    const bf16_t* Kg = Kh + hoff;
    const bf16_t* Vg = Vh + hoff;

    __syncthreads();
    if (tid < 64) { sM[tid] = -1e30f; sL[tid] = 0.0f; }

    // Q fragments for this wave's 16 rows, all of D, held in registers
    v16bf aq[4];
#pragma unroll
    for (int kk = 0; kk < 4; ++kk)
      aq[kk] = frag_a(Qg + (size_t)(s0 + qb * 16) * D_DIM + kk * 32, D_DIM, lane);

    v8f acc[4];
#pragma unroll
    for (int cb = 0; cb < 4; ++cb)
#pragma unroll
      for (int r = 0; r < 8; ++r) acc[cb][r] = 0.0f;

    for (int t = 0; t < 16; ++t) {
      __syncthreads();  // prev PV done with sVt/sP; stats init visible

      // ---- K tile: async global->LDS DMA, 16B per lane per op ----
#pragma unroll
      for (int p = 0; p < 4; ++p) {
        int cix = tid + p * 256;        // 0..1023 chunks of 8 bf16
        int r = cix >> 4, dq = (cix & 15) * 8;
        const bf16_t* gsrc = Kg + (size_t)(t * 64 + r) * D_DIM + dq;
        unsigned ldsa = (unsigned)(uintptr_t)&sK[r][dq];
        asm volatile("global_load_async_to_lds_b128 %0, %1, off"
                     :: "v"(ldsa), "v"(gsrc)
                     : "memory");
      }
      // ---- V tile: b128 reads, transpose via 2B LDS scatter ----
#pragma unroll
      for (int p = 0; p < 4; ++p) {
        int cix = tid + p * 256;
        int r = cix >> 4, dq = (cix & 15) * 8;
        uint4 raw = *(const uint4*)(Vg + (size_t)(t * 64 + r) * D_DIM + dq);
        alignas(16) bf16_t tmp[8];
        *(uint4*)tmp = raw;
#pragma unroll
        for (int j = 0; j < 8; ++j) sVt[dq + j][r] = tmp[j];
      }
      // prefetch next key tile into L2/WGP$ while we do softmax this tile
      if (t < 15) {
        const bf16_t* pk =
            Kg + (size_t)((t + 1) * 64 + (tid >> 2)) * D_DIM + (tid & 3) * 32;
        const bf16_t* pv =
            Vg + (size_t)((t + 1) * 64 + (tid >> 2)) * D_DIM + (tid & 3) * 32;
        __builtin_prefetch(pk, 0, 1);
        __builtin_prefetch(pv, 0, 1);
      }
      asm volatile("s_wait_asynccnt 0x0" ::: "memory");
      __syncthreads();

      // ---- scores: 64x64 = Q_tile @ K_tile^T (wave: 16q x 32k) ----
      v8f c0, c1;
#pragma unroll
      for (int r = 0; r < 8; ++r) { c0[r] = 0.0f; c1[r] = 0.0f; }
#pragma unroll
      for (int kk = 0; kk < 4; ++kk) {
        v16bf b0 = frag_bt(&sK[kh * 32][kk * 32], 128, lane);
        v16bf b1 = frag_bt(&sK[kh * 32 + 16][kk * 32], 128, lane);
        c0 = wmma_bf16(aq[kk], b0, c0);
        c1 = wmma_bf16(aq[kk], b1, c1);
      }
#pragma unroll
      for (int r = 0; r < 8; ++r) {
        sS[qb * 16 + r + hi * 8][kh * 32 + l]      = c0[r] * SCALE_F;
        sS[qb * 16 + r + hi * 8][kh * 32 + 16 + l] = c1[r] * SCALE_F;
      }
      __syncthreads();

      // ---- online softmax: running max / sum (4 threads per row) ----
      {
        int row = tid >> 2, seg = tid & 3;
        float mx = -1e30f;
#pragma unroll
        for (int j = 0; j < 16; ++j) mx = fmaxf(mx, sS[row][seg * 16 + j]);
        sRed[row][seg] = mx;
      }
      __syncthreads();
      if (tid < 64) {
        float mt = fmaxf(fmaxf(sRed[tid][0], sRed[tid][1]),
                         fmaxf(sRed[tid][2], sRed[tid][3]));
        float mold = sM[tid];
        float mnew = fmaxf(mold, mt);
        sM[tid]    = mnew;
        sCorr[tid] = __expf(mold - mnew);
      }
      __syncthreads();
      {
        int row = tid >> 2, seg = tid & 3;
        float mnew = sM[row];
        float sum = 0.0f;
#pragma unroll
        for (int j = 0; j < 16; ++j) {
          float p = __expf(sS[row][seg * 16 + j] - mnew);
          sP[row][seg * 16 + j] = (bf16_t)p;
          sum += p;
        }
        sRed[row][seg] = sum;
      }
      __syncthreads();
      if (tid < 64) {
        sL[tid] = sL[tid] * sCorr[tid] +
                  (sRed[tid][0] + sRed[tid][1] + sRed[tid][2] + sRed[tid][3]);
      }
      __syncthreads();

      // ---- PV: O = O*corr + P @ V_tile   (wave: 16q x 64d) ----
      float cr[8];
#pragma unroll
      for (int r = 0; r < 8; ++r) cr[r] = sCorr[qb * 16 + r + hi * 8];
#pragma unroll
      for (int cb = 0; cb < 4; ++cb)
#pragma unroll
        for (int r = 0; r < 8; ++r) acc[cb][r] *= cr[r];
#pragma unroll
      for (int ks = 0; ks < 2; ++ks) {
        v16bf a = frag_a(&sP[qb * 16][ks * 32], 64, lane);
#pragma unroll
        for (int cb = 0; cb < 4; ++cb) {
          v16bf b = frag_bt(&sVt[dh * 64 + cb * 16][ks * 32], 64, lane);
          acc[cb] = wmma_bf16(a, b, acc[cb]);
        }
      }
    }  // key tiles

    // finalize head: divide by softmax denom, accumulate 1/H of it
    float invl[8];
#pragma unroll
    for (int r = 0; r < 8; ++r) invl[r] = 1.0f / sL[qb * 16 + r + hi * 8];
#pragma unroll
    for (int cb = 0; cb < 4; ++cb)
#pragma unroll
      for (int r = 0; r < 8; ++r)
        macc[cb][r] += acc[cb][r] * invl[r] * (1.0f / (float)H_NUM);
  }  // heads

  // head-mean + residual, fp32 out [B,M,S,D]
#pragma unroll
  for (int cb = 0; cb < 4; ++cb)
#pragma unroll
    for (int r = 0; r < 8; ++r) {
      int s = s0 + qb * 16 + r + hi * 8;
      int d = dh * 64 + cb * 16 + l;
      size_t idx = ((size_t)bm * S_LEN + s) * D_DIM + d;
      out[idx] = macc[cb][r] + qin[idx];
    }
}

// ---------------------------------------------------------------------------
// Launcher
// ---------------------------------------------------------------------------
extern "C" void kernel_launch(void* const* d_in, const int* in_sizes, int n_in,
                              void* d_out, int out_size, void* d_ws,
                              size_t ws_size, hipStream_t stream) {
  (void)in_sizes; (void)n_in; (void)out_size; (void)ws_size;

  const float* q  = (const float*)d_in[0];
  const float* k  = (const float*)d_in[1];
  const float* v  = (const float*)d_in[2];
  const float* Wq = (const float*)d_in[3];
  const float* Wk = (const float*)d_in[4];
  const float* Wv = (const float*)d_in[5];
  float* out = (float*)d_out;

  const size_t head_elems = (size_t)BM_CNT * H_NUM * S_LEN * D_DIM;  // 8.39M
  bf16_t* Qh = (bf16_t*)d_ws;
  bf16_t* Kh = Qh + head_elems;
  bf16_t* Vh = Kh + head_elems;  // total 3 * 16 MB = 48 MB of workspace

  dim3 g1(E_DIM / 128, S_LEN / 64, 3 * BM_CNT);  // (8, 16, 24)
  mhsa_proj_kernel<<<g1, 256, 0, stream>>>(q, k, v, Wq, Wk, Wv, Qh, Kh, Vh);

  dim3 g2(S_LEN / 64, BM_CNT);  // (16, 8)
  mhsa_attn_kernel<<<g2, 256, 0, stream>>>(Qh, Kh, Vh, q, out);
}
